// BitLinear_25520695672955
// MI455X (gfx1250) — compile-verified, matrix operand order checked by reading, not run on hardware
//
#include <hip/hip_runtime.h>

typedef __attribute__((ext_vector_type(16))) _Float16 v16h;
typedef __attribute__((ext_vector_type(8)))  _Float16 v8h;
typedef __attribute__((ext_vector_type(4)))  _Float16 v4h;
typedef __attribute__((ext_vector_type(8)))  float    v8f;
typedef __attribute__((ext_vector_type(4)))  float    v4f;

#define K_IN   4096
#define N_OUT  11008
#define M_ROWS 128
#define W_COUNT (11008u * 4096u)   // 45,088,768

// ---------------------------------------------------------------------------
// Kernel 1: sum(|W|) reduction. float4 grid-stride loads, wave32 shuffle
// reduce, one float atomic per wave.
// ---------------------------------------------------------------------------
__global__ __launch_bounds__(256)
void bitlin_abs_sum(const float* __restrict__ w, float* __restrict__ out_sum,
                    unsigned n4) {
  unsigned tid    = blockIdx.x * blockDim.x + threadIdx.x;
  unsigned stride = gridDim.x * blockDim.x;
  const v4f* w4 = reinterpret_cast<const v4f*>(w);
  float s = 0.0f;
  for (unsigned i = tid; i < n4; i += stride) {
    v4f v = w4[i];
    s += __builtin_fabsf(v[0]) + __builtin_fabsf(v[1]) +
         __builtin_fabsf(v[2]) + __builtin_fabsf(v[3]);
  }
  #pragma unroll
  for (int off = 16; off > 0; off >>= 1)
    s += __shfl_down(s, off, 32);
  if ((threadIdx.x & 31) == 0)
    atomicAdd(out_sum, s);
}

// ---------------------------------------------------------------------------
// Kernel 2: x fp32 -> f16 (ternary weights exact in f16; accum in f32 WMMA).
// ---------------------------------------------------------------------------
__global__ __launch_bounds__(256)
void bitlin_x_to_f16(const float* __restrict__ x, _Float16* __restrict__ xh) {
  unsigned i0 = (blockIdx.x * blockDim.x + threadIdx.x) * 4u;
  v4f v = *reinterpret_cast<const v4f*>(x + i0);
  v4h h;
  h[0] = (_Float16)v[0]; h[1] = (_Float16)v[1];
  h[2] = (_Float16)v[2]; h[3] = (_Float16)v[3];
  *reinterpret_cast<v4h*>(xh + i0) = h;
}

// ---------------------------------------------------------------------------
// Kernel 3: fused ternary-quantize + GEMM via V_WMMA_F32_16X16X32_F16.
// Block = 256 threads (8 waves). Block b owns output columns [32b, 32b+32)
// (two 16-wide WMMA N-tiles). Wave w owns output rows [16w, 16w+16).
// Weight chunk (32 cols x 64 k) is quantized to ternary f16 in a
// double-buffered LDS tile once per block; ONE barrier per chunk; next
// chunk's weight global load issues before the barrier so it overlaps WMMA.
// Buffer selection is integer-offset arithmetic into a single 1-D __shared__
// array so stores stay ds_store_b128 (addrspace(3)), never flat.
// ---------------------------------------------------------------------------
#define LDS_STRIDE 72            // 64 + 8 pad halves -> conflict-free b128
#define BUF_ELEMS  (32 * LDS_STRIDE)

__device__ __forceinline__ _Float16 bitlin_q(float w, float inv_delta) {
  float t = __builtin_rintf(w * inv_delta);   // round-half-even, like jnp.round
  t = fminf(fmaxf(t, -1.0f), 1.0f);           // ternary {-1,0,1}
  return (_Float16)t;
}

__global__ __launch_bounds__(256)
void bitlin_wmma_gemm(const _Float16* __restrict__ xh,
                      const float*    __restrict__ w,
                      const float*    __restrict__ bias,
                      const float*    __restrict__ sum_ptr,
                      float*          __restrict__ out) {
  __shared__ _Float16 lds_w[2 * BUF_ELEMS];   // 2 x 4.5 KB, double buffered

  const int tid  = threadIdx.x;
  const int lane = tid & 31;
  const int wave = tid >> 5;
  const int n0   = blockIdx.x * 32;

  const float delta     = (*sum_ptr) * (1.0f / (float)W_COUNT) + 1e-8f;
  const float inv_delta = 1.0f / delta;

  // WMMA fragment geometry (wave32):
  //  A (16x32 f16): lane L -> row M = L&15; halves[0..7]=K 8g..8g+7,
  //                 halves[8..15]=K 16+8g..23+8g, g = L>>4.
  //  B (32x16 f16): lane L -> col N = L&15; same K packing as A.
  //  C (16x16 f32): lane L -> col N = L&15; vgpr r -> row M = r + 8g.
  const int nl   = lane & 15;
  const int g    = lane >> 4;
  const int mrow = wave * 16 + nl;

  // staging: thread t quantizes 8 consecutive k of one weight row
  const int st_n    = tid >> 3;               // 0..31
  const int st_k    = (tid & 7) * 8;          // 0..56
  const int st_base = st_n * LDS_STRIDE + st_k;
  const float* wrow = w + (size_t)(n0 + st_n) * K_IN + st_k;

  // per-lane B-fragment base offsets (halves) within one buffer
  const int b_base0 = nl * LDS_STRIDE + 8 * g;
  const int b_base1 = (nl + 16) * LDS_STRIDE + 8 * g;

  const _Float16* arow = xh + (size_t)mrow * K_IN;

  v8f c0 = {}, c1 = {};

  // prologue: stage chunk 0 into buffer 0
  {
    v4f w0 = *reinterpret_cast<const v4f*>(wrow);
    v4f w1 = *reinterpret_cast<const v4f*>(wrow + 4);
    v8h q;
    #pragma unroll
    for (int j = 0; j < 4; ++j) {
      q[j]     = bitlin_q(w0[j], inv_delta);
      q[j + 4] = bitlin_q(w1[j], inv_delta);
    }
    *reinterpret_cast<v8h*>(&lds_w[st_base]) = q;
  }

  for (int kc = 0; kc < 64; ++kc) {
    const int k0      = kc * 64;
    const int cur_off = (kc & 1) * BUF_ELEMS;        // integer select, LDS-only
    const int nxt_off = ((kc + 1) & 1) * BUF_ELEMS;
    const bool have_next = (kc + 1) < 64;

    // issue next chunk's weight load BEFORE the barrier (registers only)
    v4f nw0 = {}, nw1 = {};
    if (have_next) {
      nw0 = *reinterpret_cast<const v4f*>(wrow + k0 + 64);
      nw1 = *reinterpret_cast<const v4f*>(wrow + k0 + 68);
    }

    __syncthreads();   // buf[kc&1] staged; prior reads of other buffer done

    #pragma unroll
    for (int s = 0; s < 64; s += 32) {
      const _Float16* ap = arow + (k0 + s + 8 * g);
      v8h alo = *reinterpret_cast<const v8h*>(ap);
      v8h ahi = *reinterpret_cast<const v8h*>(ap + 16);
      const _Float16* bp0 = &lds_w[cur_off + b_base0 + s];
      const _Float16* bp1 = &lds_w[cur_off + b_base1 + s];
      v8h b0lo = *reinterpret_cast<const v8h*>(bp0);
      v8h b0hi = *reinterpret_cast<const v8h*>(bp0 + 16);
      v8h b1lo = *reinterpret_cast<const v8h*>(bp1);
      v8h b1hi = *reinterpret_cast<const v8h*>(bp1 + 16);
      v16h a, b0, b1;
      #pragma unroll
      for (int i = 0; i < 8; ++i) {
        a[i]  = alo[i];  a[i + 8]  = ahi[i];
        b0[i] = b0lo[i]; b0[i + 8] = b0hi[i];
        b1[i] = b1lo[i]; b1[i + 8] = b1hi[i];
      }
      c0 = __builtin_amdgcn_wmma_f32_16x16x32_f16(
               false, a, false, b0, (short)0, c0, false, false);
      c1 = __builtin_amdgcn_wmma_f32_16x16x32_f16(
               false, a, false, b1, (short)0, c1, false, false);
    }

    // quantize + store next chunk AFTER current ds_loads (DScnt in-order)
    if (have_next) {
      v8h q;
      #pragma unroll
      for (int j = 0; j < 4; ++j) {
        q[j]     = bitlin_q(nw0[j], inv_delta);
        q[j + 4] = bitlin_q(nw1[j], inv_delta);
      }
      *reinterpret_cast<v8h*>(&lds_w[nxt_off + st_base]) = q;
    }
  }

  // epilogue: out = c * delta + bias
  const int   nc0 = n0 + nl;
  const int   nc1 = n0 + 16 + nl;
  const float bv0 = bias[nc0];
  const float bv1 = bias[nc1];
  #pragma unroll
  for (int r = 0; r < 8; ++r) {
    int m = wave * 16 + r + 8 * g;
    out[(size_t)m * N_OUT + nc0] = c0[r] * delta + bv0;
    out[(size_t)m * N_OUT + nc1] = c1[r] * delta + bv1;
  }
}

// ---------------------------------------------------------------------------
extern "C" void kernel_launch(void* const* d_in, const int* in_sizes, int n_in,
                              void* d_out, int out_size, void* d_ws, size_t ws_size,
                              hipStream_t stream) {
  const float* x    = (const float*)d_in[0];   // [8,16,4096]
  const float* wgt  = (const float*)d_in[1];   // [11008,4096]
  const float* bias = (const float*)d_in[2];   // [11008]
  float* out = (float*)d_out;                  // [8,16,11008] f32

  float*    d_sum = (float*)d_ws;                          // 4 B
  _Float16* d_xh  = (_Float16*)((char*)d_ws + 256);        // 1 MB x in f16

  (void)in_sizes; (void)n_in; (void)out_size; (void)ws_size;

  hipMemsetAsync(d_ws, 0, sizeof(float), stream);

  bitlin_abs_sum<<<2048, 256, 0, stream>>>(wgt, d_sum, W_COUNT / 4);

  // 128*4096 = 524288 elems, 4 per thread -> 512 blocks of 256
  bitlin_x_to_f16<<<512, 256, 0, stream>>>(x, d_xh);

  // 11008/32 = 344 column strips
  bitlin_wmma_gemm<<<344, 256, 0, stream>>>(d_xh, wgt, bias, d_sum, out);
}
